// graph_transformer_45999099740559
// MI455X (gfx1250) — matrix operand order repaired and checked by reading, not compiled, not run on hardware
//
#include <hip/hip_runtime.h>
#include <math.h>

#define N_NODES 25000
#define N_EDGES 400000
#define D_IN 64
#define D_MODEL 128
#define D_OUT 64
#define N_HEADS 4
#define HD 32
#define N_LAYERS 2

typedef __attribute__((ext_vector_type(16))) _Float16 v16h;
typedef __attribute__((ext_vector_type(8)))  _Float16 v8h;
typedef __attribute__((ext_vector_type(8)))  float    v8f;

// ---------------------------------------------------------------------------
// Generic GEMM: C[M,N] = A[M,K] @ B[K,N] (+bias[N]) (+pos[N]), fp32 in/out,
// fp16 WMMA compute with f32 accumulation. One 16x16 tile per wave.
// Block = 32 * (N/16) threads, grid.x = ceil(M/16). Block stages:
//   As[16][K]   (row-major f16)   -- the block's 16 A rows
//   Bs[N][K]    (transposed f16)  -- full weight matrix, contiguous per column
// so each wave's fragment reads are contiguous LDS b128 loads.
// ---------------------------------------------------------------------------
__global__ void __launch_bounds__(256)
wmma_gemm_kernel(const float* __restrict__ A, const float* __restrict__ B,
                 float* __restrict__ C, const float* __restrict__ bias,
                 const float* __restrict__ pos, int M, int K, int N)
{
    extern __shared__ _Float16 smem[];
    _Float16* As = smem;            // 16 * K
    _Float16* Bs = smem + 16 * K;   // N * K (transposed)

    const int tm   = blockIdx.x;
    const int tn   = threadIdx.x >> 5;
    const int lane = threadIdx.x & 31;
    const int l    = lane & 15;
    const int half = lane >> 4;
    const int nthr = blockDim.x;

    // Stage A slab (clamp OOB rows; harmless, never stored)
    for (int idx = threadIdx.x; idx < 16 * K; idx += nthr) {
        int r = idx / K, k = idx - r * K;
        int row = tm * 16 + r; if (row >= M) row = M - 1;
        As[idx] = (_Float16)A[(size_t)row * K + k];
    }
    // Stage B transposed (global reads coalesced over n)
    for (int idx = threadIdx.x; idx < N * K; idx += nthr) {
        int k = idx / N, n = idx - k * N;
        Bs[n * K + k] = (_Float16)B[idx];
    }
    __syncthreads();

    const int col = tn * 16 + l;
    v8f acc = {};
    for (int kk = 0; kk < K; kk += 32) {
        // A 16x32 f16, k order loaded here: lane-half0 -> K kk..kk+15,
        //                     lane-half1 -> K kk+16..kk+31
        v8h a0 = *(const v8h*)(As + l * K + kk + half * 16);
        v8h a1 = *(const v8h*)(As + l * K + kk + half * 16 + 8);
        v16h a, b;
        #pragma unroll
        for (int i = 0; i < 8; ++i) { a[i] = a0[i]; a[8 + i] = a1[i]; }
        // B 32x16 f16, same k order as A: lane-half0 -> K kk..kk+15 of column l,
        //                     lane-half1 -> K kk+16..kk+31
        b = *(const v16h*)(Bs + col * K + kk + half * 16);
        acc = __builtin_amdgcn_wmma_f32_16x16x32_f16(
                  false, a, false, b, (short)0, acc, false, false);
    }

    float add = (bias ? bias[col] : 0.0f) + (pos ? pos[col] : 0.0f);
    // C/D f32 16x16 layout: VGPR r -> M = 8*half + r, N = l
    #pragma unroll
    for (int r = 0; r < 8; ++r) {
        int m = tm * 16 + half * 8 + r;
        if (m < M) C[(size_t)m * N + col] = acc[r] + add;
    }
}

// ---------------------------------------------------------------------------
// Edge pass 1: per (edge, head) -> exp(clip(q.k)) and segment-sum of norms
// ---------------------------------------------------------------------------
__global__ void edge_att_kernel(const int* __restrict__ ei,
                                const float* __restrict__ Q,
                                const float* __restrict__ Kn,
                                float* __restrict__ exp_att,
                                float* __restrict__ att_norm)
{
    int t = blockIdx.x * blockDim.x + threadIdx.x;
    if (t >= N_EDGES * N_HEADS) return;
    int e = t >> 2;
    int h = t & 3;
    int r = ei[e];
    int c = ei[N_EDGES + e];
    const float4* q = (const float4*)(Q  + (size_t)r * D_MODEL + h * HD);
    const float4* k = (const float4*)(Kn + (size_t)c * D_MODEL + h * HD);
    float dot = 0.0f;
    #pragma unroll
    for (int i = 0; i < HD / 4; ++i) {
        float4 qa = q[i], kb = k[i];
        dot += qa.x * kb.x + qa.y * kb.y + qa.z * kb.z + qa.w * kb.w;
    }
    dot = fminf(10.0f, fmaxf(-10.0f, dot));
    float ea = __expf(dot);
    exp_att[t] = ea;
    atomicAdd(att_norm + r * N_HEADS + h, ea);
}

// ---------------------------------------------------------------------------
// Edge pass 2: agg[row] += (exp_att / (norm+eps)) * V[col] (float L2 atomics)
// one thread per (edge, 4-float slice of the 128-wide feature)
// ---------------------------------------------------------------------------
__global__ void edge_agg_kernel(const int* __restrict__ ei,
                                const float* __restrict__ V,
                                const float* __restrict__ exp_att,
                                const float* __restrict__ att_norm,
                                float* __restrict__ agg)
{
    int t = blockIdx.x * blockDim.x + threadIdx.x;
    if (t >= N_EDGES * 32) return;
    int e = t >> 5;
    int j = t & 31;      // 4 floats per j -> 128 total
    int h = j >> 3;      // head = floor(4j/32)
    int r = ei[e];
    int c = ei[N_EDGES + e];
    float w = exp_att[e * N_HEADS + h] /
              (att_norm[r * N_HEADS + h] + 1e-8f);
    float4 v = *(const float4*)(V + (size_t)c * D_MODEL + j * 4);
    float* dst = agg + (size_t)r * D_MODEL + j * 4;
    atomicAdd(dst + 0, w * v.x);
    atomicAdd(dst + 1, w * v.y);
    atomicAdd(dst + 2, w * v.z);
    atomicAdd(dst + 3, w * v.w);
}

// ---------------------------------------------------------------------------
// Residual + LayerNorm, one wave32 per node (4 floats/lane), in place.
// ---------------------------------------------------------------------------
__global__ void resid_ln_kernel(float* __restrict__ embeds,
                                const float* __restrict__ agg,
                                const float* __restrict__ scale,
                                const float* __restrict__ bias)
{
    int node = (blockIdx.x * blockDim.x + threadIdx.x) >> 5;
    int lane = threadIdx.x & 31;
    if (node >= N_NODES) return;
    float* x = embeds + (size_t)node * D_MODEL;
    const float* a = agg + (size_t)node * D_MODEL;
    float4 v = *(const float4*)(x + lane * 4);
    float4 g = *(const float4*)(a + lane * 4);
    v.x += g.x; v.y += g.y; v.z += g.z; v.w += g.w;

    float s = v.x + v.y + v.z + v.w;
    #pragma unroll
    for (int off = 16; off > 0; off >>= 1) s += __shfl_xor(s, off, 32);
    float mu = s * (1.0f / D_MODEL);

    float dx = v.x - mu, dy = v.y - mu, dz = v.z - mu, dw = v.w - mu;
    float ss = dx * dx + dy * dy + dz * dz + dw * dw;
    #pragma unroll
    for (int off = 16; off > 0; off >>= 1) ss += __shfl_xor(ss, off, 32);
    float rstd = rsqrtf(ss * (1.0f / D_MODEL) + 1e-6f);

    float4 sc = *(const float4*)(scale + lane * 4);
    float4 bi = *(const float4*)(bias + lane * 4);
    float4 o;
    o.x = dx * rstd * sc.x + bi.x;
    o.y = dy * rstd * sc.y + bi.y;
    o.z = dz * rstd * sc.z + bi.z;
    o.w = dw * rstd * sc.w + bi.w;
    *(float4*)(x + lane * 4) = o;
}

__global__ void zero_kernel(float* __restrict__ p, int n)
{
    int i = blockIdx.x * blockDim.x + threadIdx.x;
    if (i < n) p[i] = 0.0f;
}

// ---------------------------------------------------------------------------
static inline void launch_gemm(const float* A, const float* B, float* C,
                               const float* bias, const float* pos,
                               int M, int K, int N, hipStream_t s)
{
    dim3 grid((M + 15) / 16);
    dim3 block(32 * (N / 16));
    size_t shmem = (size_t)(16 * K + N * K) * 2; // f16 elements
    wmma_gemm_kernel<<<grid, block, shmem, s>>>(A, B, C, bias, pos, M, K, N);
}

extern "C" void kernel_launch(void* const* d_in, const int* in_sizes, int n_in,
                              void* d_out, int out_size, void* d_ws, size_t ws_size,
                              hipStream_t stream)
{
    const float* graph_node = (const float*)d_in[0];   // [25000,64]
    const int*   edge_index = (const int*)  d_in[1];   // [2,400000]
    const float* W_P_w      = (const float*)d_in[2];   // [64,128]
    const float* W_P_b      = (const float*)d_in[3];   // [128]
    const float* W_pos      = (const float*)d_in[4];   // [1,128]
    const float* qT         = (const float*)d_in[5];   // [2,128,128]
    const float* kT         = (const float*)d_in[6];
    const float* vT         = (const float*)d_in[7];
    const float* ln_scale   = (const float*)d_in[8];   // [2,128]
    const float* ln_bias    = (const float*)d_in[9];
    const float* inv_w      = (const float*)d_in[10];  // [128,64]
    const float* inv_b      = (const float*)d_in[11];  // [64]
    float* out = (float*)d_out;

    const size_t ND = (size_t)N_NODES * D_MODEL;       // 3.2M floats
    float* ws       = (float*)d_ws;
    float* embeds   = ws;
    float* Qb       = ws + 1 * ND;
    float* Kb       = ws + 2 * ND;
    float* Vb       = ws + 3 * ND;
    float* agg      = ws + 4 * ND;
    float* att_norm = ws + 5 * ND;                      // N_NODES*N_HEADS
    float* exp_att  = ws + 5 * ND + N_NODES * N_HEADS;  // N_EDGES*N_HEADS

    // Projection + positional encoding: embeds = X @ W_P + b + pos
    launch_gemm(graph_node, W_P_w, embeds, W_P_b, W_pos,
                N_NODES, D_IN, D_MODEL, stream);

    const int zero_n = (int)ND + N_NODES * N_HEADS;     // agg + att_norm (contiguous)
    for (int l = 0; l < N_LAYERS; ++l) {
        const float* qw = qT + (size_t)l * D_MODEL * D_MODEL;
        const float* kw = kT + (size_t)l * D_MODEL * D_MODEL;
        const float* vw = vT + (size_t)l * D_MODEL * D_MODEL;

        // Node-level Q/K/V (algebraic hoist of edge-level matmuls)
        launch_gemm(embeds, qw, Qb, nullptr, nullptr, N_NODES, D_MODEL, D_MODEL, stream);
        launch_gemm(embeds, kw, Kb, nullptr, nullptr, N_NODES, D_MODEL, D_MODEL, stream);
        launch_gemm(embeds, vw, Vb, nullptr, nullptr, N_NODES, D_MODEL, D_MODEL, stream);

        zero_kernel<<<(zero_n + 255) / 256, 256, 0, stream>>>(agg, zero_n);

        edge_att_kernel<<<(N_EDGES * N_HEADS + 255) / 256, 256, 0, stream>>>(
            edge_index, Qb, Kb, exp_att, att_norm);

        edge_agg_kernel<<<(N_EDGES * 32 + 255) / 256, 256, 0, stream>>>(
            edge_index, Vb, exp_att, att_norm, agg);

        resid_ln_kernel<<<(N_NODES * 32 + 255) / 256, 256, 0, stream>>>(
            embeds, agg, ln_scale + l * D_MODEL, ln_bias + l * D_MODEL);
    }

    // Output projection: out = embeds @ inv_w + inv_b
    launch_gemm(embeds, inv_w, out, inv_b, nullptr,
                N_NODES, D_MODEL, D_OUT, stream);
}
